// ExtendedKalmanFilterWithIK_24257975287974
// MI455X (gfx1250) — matrix operand order repaired and checked by reading, not compiled, not run on hardware
//
#include <hip/hip_runtime.h>

// ---------------------------------------------------------------------------
// Extended Kalman Filter with NN measurement model, CDNA5 (gfx1250) WMMA path.
//   B=2048, M=6 (state), N=3 (meas), T=256 (sequential), HID=64.
// One wave32 per 16-batch tile (128 blocks x 32 threads).
// Primal + 3 JVP tangents fused into a 64x64 matrix -> v_wmma_f32_16x16x32_f16.
// Branch-free tanh (hw v_tanh_f32 if available, else exp2+rcp) to keep the
// serial scan free of divergent EXEC branches.
// ---------------------------------------------------------------------------

typedef __attribute__((ext_vector_type(16))) _Float16 v16h;
typedef __attribute__((ext_vector_type(8)))  _Float16 v8h;
typedef __attribute__((ext_vector_type(8)))  float    v8f;

#define WMMA(A, B, C) \
  __builtin_amdgcn_wmma_f32_16x16x32_f16(false, (A), false, (B), (short)0, (C), false, false)

constexpr int BSZ = 2048;
constexpr int MD  = 6;
constexpr int ND  = 3;
constexpr int TT  = 256;
constexpr int HID = 64;
constexpr int TB  = 16;     // batch tile per wave
constexpr int APITCH = 72;  // LDS row pitch in halves (pad to dodge bank conflicts)

// Branch-free tanh: prefer the CDNA5 hardware transcendental, else use
// tanh(x) = sign(x) * (1 - 2/(exp2(2*log2e*|x|) + 1)) with v_exp_f32/v_rcp_f32.
__device__ __forceinline__ float fast_tanh(float x) {
#if __has_builtin(__builtin_amdgcn_tanhf)
  return __builtin_amdgcn_tanhf(x);
#else
  const float ax = __builtin_fabsf(x);
  const float e  = __builtin_amdgcn_exp2f(ax * 2.885390081777927f); // 2*log2(e)
  const float r  = 1.0f - 2.0f * __builtin_amdgcn_rcpf(e + 1.0f);
  return __builtin_copysignf(r, x);
#endif
}

__device__ __forceinline__ v16h zero16h() {
  v16h z;
#pragma unroll
  for (int i = 0; i < 16; ++i) z[i] = (_Float16)0.f;
  return z;
}

// Load one 16x32 f16 A-fragment for the next layer from row-major LDS.
// lane 0-15 hold M rows with halves = K{0..7},{16..23}; lanes 16-31 K{8..15},{24..31}.
__device__ __forceinline__ v16h load_afrag(const _Float16* rowp) {
  v8h lo = *(const v8h*)(rowp);
  v8h hi = *(const v8h*)(rowp + 16);
  return __builtin_shufflevector(lo, hi, 0, 1, 2, 3, 4, 5, 6, 7,
                                 8, 9, 10, 11, 12, 13, 14, 15);
}

// Store one 16x16 f32 C tile (WMMA accumulator layout) into LDS as f16.
// C layout: lane = N col (mod 16), VGPR v -> row v + 8*(lane>>4).
__device__ __forceinline__ void store_tile_f16(_Float16 (*act)[APITCH], int mt, int nt,
                                               v8f c, int n, int hs) {
#pragma unroll
  for (int v = 0; v < 8; ++v)
    act[mt * 16 + hs * 8 + v][nt * 16 + n] = (_Float16)c[v];
}

__global__ __launch_bounds__(32)
void ekf_ik_wmma_kernel(const float* __restrict__ y,      // (B, N, T)
                        const float* __restrict__ m1x0,   // (B, M, 1)
                        const float* __restrict__ m2x0,   // (B, M, M)
                        const float* __restrict__ Qm,     // (M, M)
                        const float* __restrict__ Rm,     // (N, N)
                        const float* __restrict__ W1,     // (HID, N)
                        const float* __restrict__ b1,     // (HID,)
                        const float* __restrict__ W2,     // (HID, HID)
                        const float* __restrict__ b2,     // (HID,)
                        const float* __restrict__ W3,     // (N, HID)
                        const float* __restrict__ b3,     // (N,)
                        float* __restrict__ xo,           // (B, M, T)
                        float* __restrict__ so,           // (B, M, M, T)
                        float* __restrict__ ko)           // (B, M, N, T)
{
  __shared__ __align__(16) _Float16 actA[64][APITCH];  // layer-1 outputs (primal+tangents)
  __shared__ __align__(16) _Float16 actB[64][APITCH];  // layer-2 outputs
  __shared__ __align__(16) float    outL[4][16][16];   // layer-3 outputs (f32)
  __shared__ float QR[36 + 9];                         // Q then R staged in LDS

  const int lane = threadIdx.x;
  const int n    = lane & 15;
  const int hs   = lane >> 4;
  const int b0   = blockIdx.x * TB;

  // ---- stage Q, R into LDS ------------------------------------------------
  for (int i = lane; i < 36; i += 32) QR[i] = Qm[i];
  for (int i = lane; i < 9; i += 32)  QR[36 + i] = Rm[i];

  // ---- resident f16 weight fragments (constant across all 256 steps) ------
  // B2f[kt][nt]: 32x16 tile of W2^T; lane = out col j=nt*16+n, halves = K.
  v16h B2f[2][4];
#pragma unroll
  for (int kt = 0; kt < 2; ++kt)
#pragma unroll
    for (int nt = 0; nt < 4; ++nt) {
      const float* wp = W2 + (nt * 16 + n) * HID + kt * 32 + hs * 8;
#pragma unroll
      for (int h = 0; h < 16; ++h) B2f[kt][nt][h] = (_Float16)wp[h < 8 ? h : h + 8];
    }
  // B1f[nt]: 32x16 tile of W1^T; only K=0..2 nonzero (live in lanes 0-15).
  v16h B1f[4];
#pragma unroll
  for (int nt = 0; nt < 4; ++nt) {
    B1f[nt] = zero16h();
    if (lane < 16) {
#pragma unroll
      for (int k = 0; k < 3; ++k)
        B1f[nt][k] = (_Float16)W1[(nt * 16 + n) * ND + k];
    }
  }
  // B3f[kt]: 32x16 tile of W3^T; only cols i<3 valid.
  v16h B3f[2];
#pragma unroll
  for (int kt = 0; kt < 2; ++kt) {
    B3f[kt] = zero16h();
    if (n < 3) {
      const float* wp = W3 + n * HID + kt * 32 + hs * 8;
#pragma unroll
      for (int h = 0; h < 16; ++h) B3f[kt][h] = (_Float16)wp[h < 8 ? h : h + 8];
    }
  }
  // Per-lane biases (col j = nt*16 + n) and layer-1 tangent columns of W1.
  float b1v[4], b2v[4], w1c[3][4];
#pragma unroll
  for (int nt = 0; nt < 4; ++nt) {
    b1v[nt] = b1[nt * 16 + n];
    b2v[nt] = b2[nt * 16 + n];
#pragma unroll
    for (int k = 0; k < 3; ++k) w1c[k][nt] = W1[(nt * 16 + n) * ND + k];
  }
  const float b3v[3] = {b3[0], b3[1], b3[2]};
  const float LOc[6] = {-0.8f, -0.25f, -0.2f, -0.5f, -0.5f, -0.5f};
  const float HIc[6] = { 0.8f,  0.25f,  0.55f, 0.5f,  0.5f,  0.5f};

  // ---- per-lane filter state (lane b < 16 owns batch b0+b) ----------------
  float m1x[6];
  float P[36];
  if (lane < 16) {
    const int b = b0 + lane;
#pragma unroll
    for (int m = 0; m < 6; ++m) m1x[m] = m1x0[b * 6 + m];
#pragma unroll
    for (int i = 0; i < 36; ++i) P[i] = m2x0[b * 36 + i];
  }
  __syncthreads();

  // =========================== time scan ===================================
  for (int t = 0; t < TT; ++t) {
    // ---- build A1 (16x32 f16): rows = batch, cols 0..2 = pos, rest 0 ------
    v16h a1 = zero16h();
    if (lane < 16) {
      a1[0] = (_Float16)m1x[0];
      a1[1] = (_Float16)m1x[1];
      a1[2] = (_Float16)m1x[2];
    }

    // ---- layer 1: primal via WMMA, tangents analytic ----------------------
    v8f d1[4];
#pragma unroll
    for (int nt = 0; nt < 4; ++nt) {
      v8f c = {};
      c = WMMA(a1, B1f[nt], c);
      v8f h, d;
#pragma unroll
      for (int v = 0; v < 8; ++v) {
        float hv = fast_tanh(c[v] + b1v[nt]);
        h[v] = hv;
        d[v] = 1.f - hv * hv;
      }
      store_tile_f16(actA, 0, nt, h, n, hs);
      d1[nt] = d;
    }
    // tangent rows of layer 1: t1[b,j] = (1-h1^2)[b,j] * W1[j,k]
#pragma unroll
    for (int k = 0; k < 3; ++k)
#pragma unroll
      for (int nt = 0; nt < 4; ++nt) {
        v8f tt;
#pragma unroll
        for (int v = 0; v < 8; ++v) tt[v] = d1[nt][v] * w1c[k][nt];
        store_tile_f16(actA, k + 1, nt, tt, n, hs);
      }
    __syncthreads();

    // ---- layer 2: 64x64 @ 64x64 via 32 WMMAs ------------------------------
    v8f d2[4];
#pragma unroll
    for (int mt = 0; mt < 4; ++mt) {
      v16h a0 = load_afrag(&actA[mt * 16 + n][0 * 32 + hs * 8]);
      v16h a1k = load_afrag(&actA[mt * 16 + n][1 * 32 + hs * 8]);
#pragma unroll
      for (int nt = 0; nt < 4; ++nt) {
        v8f c = {};
        c = WMMA(a0, B2f[0][nt], c);
        c = WMMA(a1k, B2f[1][nt], c);
        if (mt == 0) {
          v8f h, d;
#pragma unroll
          for (int v = 0; v < 8; ++v) {
            float hv = fast_tanh(c[v] + b2v[nt]);
            h[v] = hv;
            d[v] = 1.f - hv * hv;
          }
          store_tile_f16(actB, 0, nt, h, n, hs);
          d2[nt] = d;
        } else {
          v8f tt;
#pragma unroll
          for (int v = 0; v < 8; ++v) tt[v] = d2[nt][v] * c[v];
          store_tile_f16(actB, mt, nt, tt, n, hs);
        }
      }
    }
    __syncthreads();

    // ---- layer 3: 64x64 @ 64x16(pad of 3) via 8 WMMAs ---------------------
#pragma unroll
    for (int mt = 0; mt < 4; ++mt) {
      v16h a0 = load_afrag(&actB[mt * 16 + n][0 * 32 + hs * 8]);
      v16h a1k = load_afrag(&actB[mt * 16 + n][1 * 32 + hs * 8]);
      v8f c = {};
      c = WMMA(a0, B3f[0], c);
      c = WMMA(a1k, B3f[1], c);
#pragma unroll
      for (int v = 0; v < 8; ++v) outL[mt][hs * 8 + v][n] = c[v];
    }
    __syncthreads();

    // ---- scalar EKF update (lanes 0-15, one batch each) -------------------
    if (lane < 16) {
      const int b = b0 + lane;

      // P_prior = P + Q
      float Pp[36];
#pragma unroll
      for (int i = 0; i < 36; ++i) Pp[i] = P[i] + QR[i];

      // m1y, H (Jacobian) from layer-3 tiles
      float m1y[3], H[3][3];
#pragma unroll
      for (int i = 0; i < 3; ++i) m1y[i] = outL[0][lane][i] + b3v[i] + m1x[3 + i];
#pragma unroll
      for (int k = 0; k < 3; ++k)
#pragma unroll
        for (int i = 0; i < 3; ++i) H[i][k] = outL[k + 1][lane][i];

      // S = H * Pp3 * H^T + R
      float HP[3][3];
#pragma unroll
      for (int i = 0; i < 3; ++i)
#pragma unroll
        for (int j = 0; j < 3; ++j) {
          float acc = 0.f;
#pragma unroll
          for (int k = 0; k < 3; ++k) acc += H[i][k] * Pp[k * 6 + j];
          HP[i][j] = acc;
        }
      float S[3][3];
#pragma unroll
      for (int i = 0; i < 3; ++i)
#pragma unroll
        for (int j = 0; j < 3; ++j) {
          float acc = QR[36 + i * 3 + j];
#pragma unroll
          for (int k = 0; k < 3; ++k) acc += HP[i][k] * H[j][k];
          S[i][j] = acc;
        }

      // 3x3 inverse via adjugate
      float c00 = S[1][1] * S[2][2] - S[1][2] * S[2][1];
      float c01 = S[1][2] * S[2][0] - S[1][0] * S[2][2];
      float c02 = S[1][0] * S[2][1] - S[1][1] * S[2][0];
      float det = S[0][0] * c00 + S[0][1] * c01 + S[0][2] * c02;
      float rdet = 1.f / det;
      float Si[3][3];
      Si[0][0] = c00 * rdet;
      Si[1][0] = c01 * rdet;
      Si[2][0] = c02 * rdet;
      Si[0][1] = (S[0][2] * S[2][1] - S[0][1] * S[2][2]) * rdet;
      Si[1][1] = (S[0][0] * S[2][2] - S[0][2] * S[2][0]) * rdet;
      Si[2][1] = (S[0][1] * S[2][0] - S[0][0] * S[2][1]) * rdet;
      Si[0][2] = (S[0][1] * S[1][2] - S[0][2] * S[1][1]) * rdet;
      Si[1][2] = (S[0][2] * S[1][0] - S[0][0] * S[1][2]) * rdet;
      Si[2][2] = (S[0][0] * S[1][1] - S[0][1] * S[1][0]) * rdet;

      // K = Pp3 * H^T * S^-1
      float PHt[3][3];
#pragma unroll
      for (int i = 0; i < 3; ++i)
#pragma unroll
        for (int j = 0; j < 3; ++j) {
          float acc = 0.f;
#pragma unroll
          for (int k = 0; k < 3; ++k) acc += Pp[i * 6 + k] * H[j][k];
          PHt[i][j] = acc;
        }
      float K[3][3];
#pragma unroll
      for (int i = 0; i < 3; ++i)
#pragma unroll
        for (int j = 0; j < 3; ++j) {
          float acc = 0.f;
#pragma unroll
          for (int k = 0; k < 3; ++k) acc += PHt[i][k] * Si[k][j];
          K[i][j] = acc;
        }

      // innovation + state update + clip
      float dy[3];
#pragma unroll
      for (int i = 0; i < 3; ++i)
        dy[i] = y[(b * ND + i) * TT + t] - m1y[i];
#pragma unroll
      for (int i = 0; i < 3; ++i) {
        float acc = m1x[i];
#pragma unroll
        for (int j = 0; j < 3; ++j) acc += K[i][j] * dy[j];
        m1x[i] = fminf(fmaxf(acc, LOc[i]), HIc[i]);
      }
#pragma unroll
      for (int i = 3; i < 6; ++i)
        m1x[i] = fminf(fmaxf(m1x[i], LOc[i]), HIc[i]);

      // covariance update: top-left 3x3 -= K * S * K^T
      float KS[3][3];
#pragma unroll
      for (int i = 0; i < 3; ++i)
#pragma unroll
        for (int j = 0; j < 3; ++j) {
          float acc = 0.f;
#pragma unroll
          for (int k = 0; k < 3; ++k) acc += K[i][k] * S[k][j];
          KS[i][j] = acc;
        }
#pragma unroll
      for (int i = 0; i < 3; ++i)
#pragma unroll
        for (int j = 0; j < 3; ++j) {
          float acc = 0.f;
#pragma unroll
          for (int k = 0; k < 3; ++k) acc += KS[i][k] * K[j][k];
          Pp[i * 6 + j] -= acc;
        }
#pragma unroll
      for (int i = 0; i < 36; ++i) P[i] = Pp[i];

      // ---- outputs for this timestep ----
#pragma unroll
      for (int m = 0; m < 6; ++m)
        xo[(b * 6 + m) * TT + t] = m1x[m];
#pragma unroll
      for (int i = 0; i < 36; ++i)
        so[(b * 36 + i) * TT + t] = P[i];
#pragma unroll
      for (int i = 0; i < 3; ++i)
#pragma unroll
        for (int j = 0; j < 3; ++j)
          ko[(b * 18 + i * 3 + j) * TT + t] = K[i][j];
#pragma unroll
      for (int i = 3; i < 6; ++i)
#pragma unroll
        for (int j = 0; j < 3; ++j)
          ko[(b * 18 + i * 3 + j) * TT + t] = 0.f;

      // prefetch next timestep's measurement (global_prefetch_b8)
      if (t + 1 < TT) __builtin_prefetch(&y[(b * ND) * TT + t + 1], 0, 0);
    }
    __syncthreads();  // protect actA/outL reuse next iteration
  }
}

// ---------------------------------------------------------------------------
// Host launcher. Inputs in setup_inputs() order:
//  0:y 1:m1x_0 2:m2x_0 3:Q 4:R 5:W1 6:b1 7:W2 8:b2 9:W3 10:b3
// d_out = concat(x (B,M,T), sigma (B,M,M,T), KG (B,M,N,T)) as float32.
// ---------------------------------------------------------------------------
extern "C" void kernel_launch(void* const* d_in, const int* in_sizes, int n_in,
                              void* d_out, int out_size, void* d_ws, size_t ws_size,
                              hipStream_t stream) {
  const float* y    = (const float*)d_in[0];
  const float* m1x0 = (const float*)d_in[1];
  const float* m2x0 = (const float*)d_in[2];
  const float* Qm   = (const float*)d_in[3];
  const float* Rm   = (const float*)d_in[4];
  const float* W1   = (const float*)d_in[5];
  const float* b1   = (const float*)d_in[6];
  const float* W2   = (const float*)d_in[7];
  const float* b2   = (const float*)d_in[8];
  const float* W3   = (const float*)d_in[9];
  const float* b3   = (const float*)d_in[10];

  float* out = (float*)d_out;
  float* xo = out;                                 // B*M*T
  float* so = xo + (size_t)BSZ * MD * TT;          // B*M*M*T
  float* ko = so + (size_t)BSZ * MD * MD * TT;     // B*M*N*T

  dim3 grid(BSZ / TB), block(32);
  ekf_ik_wmma_kernel<<<grid, block, 0, stream>>>(y, m1x0, m2x0, Qm, Rm,
                                                 W1, b1, W2, b2, W3, b3,
                                                 xo, so, ko);
}
